// LSTMODE_55224689492309
// MI455X (gfx1250) — matrix operand order repaired and checked by reading, not compiled, not run on hardware
//
#include <hip/hip_runtime.h>
#include <hip/hip_bf16.h>
#include <math.h>

typedef _Float16 half_t;
typedef __attribute__((ext_vector_type(8)))  _Float16 v8h;
typedef __attribute__((ext_vector_type(16))) _Float16 v16h;
typedef __attribute__((ext_vector_type(8)))  float    v8f;

#define T_SEQ    384
#define BATCH    256
#define DXC      64
#define DZC      256
#define ROWS     16            // batch rows per workgroup (= WMMA M)
#define NWG      (BATCH / ROWS)
#define NTHREADS 256           // 8 wave32

// ---- dynamic LDS layout (160 KB total; gfx1250 allows 320 KB/WG) ----
#define OFF_Y     0            // 16x512 f32 : y = [h | c]
#define OFF_YSUM  32768        // 16x512 f32 : RK4 accumulator  (gates overlay spans YSUM+KOUT)
#define OFF_KOUT  65536        // 16x512 f32 : ode_f output k
#define OFF_AF16  98304        // 16x1024 f16: A operand staging / h2
#define OFF_H1    131072       // 16x1024 f16: h1
#define LDS_BYTES 163840

// ---- workspace layout (bytes) ----
#define WS_P     0                         // 64x256 f32  : (B B^T)^-1 B
#define WS_BSUM  65536                     // 1024 f32    : b_ih + b_hh
#define WS_W0H   69632                     // 1024x512 f16
#define WS_W1H   (WS_W0H + 1048576)        // 1024x1024 f16
#define WS_W2H   (WS_W1H + 2097152)        // 512x1024 f16
#define WS_WHH   (WS_W2H + 1048576)        // 1024x256 f16

__device__ __forceinline__ float sigmf(float x) { return 1.0f / (1.0f + __expf(-x)); }

// ---------------------------------------------------------------------------
// out(16 x N) = A(16 x K) @ W^T  (+bias), W row-major (N x K) f16 in global.
// TANH_F16: write tanh(.) as f16 (stride N). else: write f32 (stride N).
// One WG = 8 waves; wave w owns n-tiles w, w+8, ...  K % 32 == 0, N % 128 == 0.
// ---------------------------------------------------------------------------
template<int K, int N, bool TANH_F16>
__device__ __forceinline__ void gemm16(const half_t* __restrict__ A,     // LDS
                                       const half_t* __restrict__ W,     // global
                                       const float*  __restrict__ bias,  // global [N]
                                       void* __restrict__ outp,          // LDS
                                       int tid)
{
  const int wave = tid >> 5;
  const int lane = tid & 31;
  const int r    = lane & 15;
  const int hi   = lane >> 4;
  // A fragment: lanes 0-15 -> M=r, K blocks {0..7}+{16..23}; lanes 16-31 -> {8..15}+{24..31}
  const half_t* pa = A + r * K + hi * 8;
  for (int nt = wave; nt < (N >> 4); nt += (NTHREADS / 32)) {
    const int n = (nt << 4) + r;
    // B fragment: lane holds column n (= W row n); lanes 0-15: K {0..7}+{16..23}, lanes 16-31: {8..15}+{24..31}
    const half_t* pb = W + (size_t)n * K + hi * 8;
    v8f acc = {};
    for (int k0 = 0; k0 < K; k0 += 32) {
      v8h alo = *(const v8h*)(pa + k0);
      v8h ahi = *(const v8h*)(pa + k0 + 16);
      v16h a = __builtin_shufflevector(alo, ahi, 0,1,2,3,4,5,6,7,8,9,10,11,12,13,14,15);
      v8h blo = *(const v8h*)(pb + k0);
      v8h bhi = *(const v8h*)(pb + k0 + 16);
      v16h b = __builtin_shufflevector(blo, bhi, 0,1,2,3,4,5,6,7,8,9,10,11,12,13,14,15);
      acc = __builtin_amdgcn_wmma_f32_16x16x32_f16(false, a, false, b,
                                                   (short)0, acc, false, false);
    }
    const float bv = bias[n];
    if (TANH_F16) {
      half_t* o = (half_t*)outp;
#pragma unroll
      for (int j = 0; j < 8; ++j) {
        const int m = (hi << 3) + j;               // C/D layout: VGPR j -> M = j + 8*hi
        o[m * N + n] = (half_t)tanhf(acc[j] + bv);
      }
    } else {
      float* o = (float*)outp;
#pragma unroll
      for (int j = 0; j < 8; ++j) {
        const int m = (hi << 3) + j;
        o[m * N + n] = acc[j] + bv;
      }
    }
  }
}

// ode_f: af(16x512 f16) -> kout(16x512 f32).  af is clobbered (reused for h2).
__device__ __forceinline__ void ode_f(char* smem,
                                      const half_t* W0h, const half_t* W1h, const half_t* W2h,
                                      const float* b0, const float* b1, const float* b2, int tid)
{
  half_t* af = (half_t*)(smem + OFF_AF16);
  half_t* h1 = (half_t*)(smem + OFF_H1);
  float*  ko = (float*)(smem + OFF_KOUT);
  gemm16<512, 1024, true >(af, W0h, b0, h1, tid);  __syncthreads();
  gemm16<1024,1024, true >(h1, W1h, b1, af, tid);  __syncthreads();
  gemm16<1024, 512, false>(af, W2h, b2, ko, tid);  __syncthreads();
}

// z = x[:,t] @ P  (16x64 @ 64x256) -> yv[:, :256], c part zeroed
__device__ __forceinline__ void project_reset(float* yv, const float* __restrict__ x,
                                              const float* __restrict__ P,
                                              int wg, int t, int tid)
{
  for (int e = tid; e < ROWS * DZC; e += NTHREADS) {
    const int r = e >> 8, j = e & 255;
    const float* xr = x + ((size_t)(wg * ROWS + r) * T_SEQ + t) * DXC;
    float s = 0.f;
#pragma unroll 8
    for (int k = 0; k < DXC; ++k) s += xr[k] * P[k * DZC + j];
    yv[r * 512 + j]       = s;
    yv[r * 512 + DZC + j] = 0.f;
  }
}

// ---------------------------------------------------------------------------
// Main persistent kernel: one WG owns 16 batch rows for the whole scan.
// ---------------------------------------------------------------------------
__global__ void __launch_bounds__(NTHREADS)
lstm_ode_main(const float* __restrict__ x, const int* __restrict__ pn, const int* __restrict__ pts,
              const float* __restrict__ b0, const float* __restrict__ b1, const float* __restrict__ b2,
              const float* __restrict__ P, const float* __restrict__ bsum,
              const half_t* __restrict__ W0h, const half_t* __restrict__ W1h,
              const half_t* __restrict__ W2h, const half_t* __restrict__ Whh,
              float* __restrict__ out)
{
  extern __shared__ char smem[];
  float*  yv    = (float*)(smem + OFF_Y);
  float*  ysum  = (float*)(smem + OFF_YSUM);
  float*  ko    = (float*)(smem + OFF_KOUT);
  float*  gates = (float*)(smem + OFF_YSUM);   // 16x1024 overlay over YSUM+KOUT
  half_t* af    = (half_t*)(smem + OFF_AF16);

  const int tid  = threadIdx.x;
  const int wg   = blockIdx.x;
  const int nres = *pn;
  const int ts   = *pts;
  const float dt = (float)T_SEQ / (float)(T_SEQ * ts - 1);

  // initial carry: h0 = x[:,0] @ P, c0 = 0
  project_reset(yv, x, P, wg, 0, tid);
  __syncthreads();

  for (int t = 0; t < T_SEQ; ++t) {
    // ---- (ts-1) RK4 steps on y = [h | c] ----
    for (int s = 0; s < ts - 1; ++s) {
      for (int e = tid; e < ROWS * 512; e += NTHREADS) af[e] = (half_t)yv[e];
      __syncthreads();
      ode_f(smem, W0h, W1h, W2h, b0, b1, b2, tid);                    // k1
      for (int e = tid; e < ROWS * 512; e += NTHREADS) {
        const float k1 = ko[e];
        ysum[e] = k1;
        af[e]   = (half_t)(yv[e] + 0.5f * dt * k1);
      }
      __syncthreads();
      ode_f(smem, W0h, W1h, W2h, b0, b1, b2, tid);                    // k2
      for (int e = tid; e < ROWS * 512; e += NTHREADS) {
        const float k2 = ko[e];
        ysum[e] += 2.0f * k2;
        af[e]    = (half_t)(yv[e] + 0.5f * dt * k2);
      }
      __syncthreads();
      ode_f(smem, W0h, W1h, W2h, b0, b1, b2, tid);                    // k3
      for (int e = tid; e < ROWS * 512; e += NTHREADS) {
        const float k3 = ko[e];
        ysum[e] += 2.0f * k3;
        af[e]    = (half_t)(yv[e] + dt * k3);
      }
      __syncthreads();
      ode_f(smem, W0h, W1h, W2h, b0, b1, b2, tid);                    // k4
      for (int e = tid; e < ROWS * 512; e += NTHREADS)
        yv[e] += dt * (1.0f / 6.0f) * (ysum[e] + ko[e]);
      __syncthreads();
    }

    // ---- reset ----
    if (nres > 0 && (t % nres) == 0 && t != 0) {
      project_reset(yv, x, P, wg, t, tid);
      __syncthreads();
    }

    // ---- gates = h @ Whh^T + (b_ih + b_hh) ----
    for (int e = tid; e < ROWS * DZC; e += NTHREADS) {
      const int r = e >> 8, j = e & 255;
      af[r * DZC + j] = (half_t)yv[r * 512 + j];
    }
    __syncthreads();
    gemm16<256, 1024, false>(af, Whh, bsum, gates, tid);
    __syncthreads();

    // ---- LSTM cell + output ----
    for (int e = tid; e < ROWS * DZC; e += NTHREADS) {
      const int r = e >> 8, j = e & 255;
      const float ig = gates[r * 1024 + j];
      const float fg = gates[r * 1024 + 256 + j];
      const float gg = gates[r * 1024 + 512 + j];
      const float og = gates[r * 1024 + 768 + j];
      const float c  = yv[r * 512 + DZC + j];
      const float cn = sigmf(fg) * c + sigmf(ig) * tanhf(gg);
      const float hn = sigmf(og) * tanhf(cn);
      yv[r * 512 + j]       = hn;
      yv[r * 512 + DZC + j] = cn;
      out[((size_t)(wg * ROWS + r) * T_SEQ + t) * DZC + j] = hn;
    }
    __syncthreads();
  }
}

// ---------------------------------------------------------------------------
// Prepass 1: P = (B B^T)^{-1} B  via Gauss-Jordan on the SPD 64x64 Gram matrix
// ---------------------------------------------------------------------------
__global__ void __launch_bounds__(NTHREADS)
prep_pinv(const float* __restrict__ B, float* __restrict__ P)
{
  __shared__ float G [64][64];
  __shared__ float GI[64][64];
  __shared__ float fac[64];
  const int tid = threadIdx.x;

  for (int e = tid; e < 4096; e += NTHREADS) {
    const int i = e >> 6, j = e & 63;
    float s = 0.f;
    for (int k = 0; k < 256; ++k) s += B[i * 256 + k] * B[j * 256 + k];
    G[i][j]  = s;
    GI[i][j] = (i == j) ? 1.f : 0.f;
  }
  __syncthreads();

  for (int p = 0; p < 64; ++p) {
    const float inv = 1.0f / G[p][p];
    for (int j = tid; j < 64; j += NTHREADS) { G[p][j] *= inv; GI[p][j] *= inv; }
    __syncthreads();
    for (int r = tid; r < 64; r += NTHREADS) fac[r] = G[r][p];
    __syncthreads();
    for (int e = tid; e < 4096; e += NTHREADS) {
      const int r = e >> 6, j = e & 63;
      if (r != p) {
        G[r][j]  -= fac[r] * G[p][j];
        GI[r][j] -= fac[r] * GI[p][j];
      }
    }
    __syncthreads();
  }

  for (int e = tid; e < 64 * 256; e += NTHREADS) {
    const int i = e >> 8, j = e & 255;
    float s = 0.f;
    for (int k = 0; k < 64; ++k) s += GI[i][k] * B[k * 256 + j];
    P[e] = s;
  }
}

// ---------------------------------------------------------------------------
// Prepass 2: f32 -> f16 weight conversion (+ bsum = b_ih + b_hh)
// ---------------------------------------------------------------------------
__global__ void __launch_bounds__(NTHREADS)
prep_cvt(const float* __restrict__ W0, const float* __restrict__ W1,
         const float* __restrict__ W2, const float* __restrict__ Whh,
         const float* __restrict__ bih, const float* __restrict__ bhh,
         half_t* __restrict__ W0h, half_t* __restrict__ W1h,
         half_t* __restrict__ W2h, half_t* __restrict__ Whhh,
         float* __restrict__ bsum)
{
  const size_t i = (size_t)blockIdx.x * NTHREADS + threadIdx.x;
  if (i < 524288)  W0h[i]  = (half_t)W0[i];
  if (i < 1048576) W1h[i]  = (half_t)W1[i];
  if (i < 524288)  W2h[i]  = (half_t)W2[i];
  if (i < 262144)  Whhh[i] = (half_t)Whh[i];
  if (i < 1024)    bsum[i] = bih[i] + bhh[i];
}

// ---------------------------------------------------------------------------
extern "C" void kernel_launch(void* const* d_in, const int* in_sizes, int n_in,
                              void* d_out, int out_size, void* d_ws, size_t ws_size,
                              hipStream_t stream)
{
  const float* x   = (const float*)d_in[0];
  const int*   pn  = (const int*)  d_in[1];
  const int*   pts = (const int*)  d_in[2];
  const float* B   = (const float*)d_in[3];
  const float* W0  = (const float*)d_in[4];
  const float* b0  = (const float*)d_in[5];
  const float* W1  = (const float*)d_in[6];
  const float* b1  = (const float*)d_in[7];
  const float* W2  = (const float*)d_in[8];
  const float* b2  = (const float*)d_in[9];
  /* d_in[10] = W_ih : unused by the reference math */
  const float* Whh = (const float*)d_in[11];
  const float* bih = (const float*)d_in[12];
  const float* bhh = (const float*)d_in[13];
  float* out = (float*)d_out;

  char* ws = (char*)d_ws;
  float*  P    = (float*) (ws + WS_P);
  float*  bsum = (float*) (ws + WS_BSUM);
  half_t* W0h  = (half_t*)(ws + WS_W0H);
  half_t* W1h  = (half_t*)(ws + WS_W1H);
  half_t* W2h  = (half_t*)(ws + WS_W2H);
  half_t* Whhh = (half_t*)(ws + WS_WHH);

  // opt in to >64KB dynamic LDS (no-op / ignored error if unsupported)
  (void)hipFuncSetAttribute((const void*)lstm_ode_main,
                            hipFuncAttributeMaxDynamicSharedMemorySize, LDS_BYTES);

  prep_pinv<<<1, NTHREADS, 0, stream>>>(B, P);
  prep_cvt<<<1048576 / NTHREADS, NTHREADS, 0, stream>>>(W0, W1, W2, Whh, bih, bhh,
                                                        W0h, W1h, W2h, Whhh, bsum);
  lstm_ode_main<<<NWG, NTHREADS, LDS_BYTES, stream>>>(x, pn, pts, b0, b1, b2,
                                                      P, bsum, W0h, W1h, W2h, Whhh, out);
}
